// Conv2D_80796924772741
// MI455X (gfx1250) — 8-bit matrix operand order repaired and checked by reading, compiled, not run on hardware
//
#include <hip/hip_runtime.h>
#include <hip/hip_bf16.h>

typedef __attribute__((ext_vector_type(8))) int v8i;

#define FCH  64
#define HIN  514
#define WIN  514
#define DIN  4
#define HOUT 512
#define WOUT 512

#define Q_MANTISSA 1305500416
#define EXPONENT   (-7)
#define OUT_ZP     (-5)


__device__ __forceinline__ unsigned pack_i4(const int4 q) {
    return ((unsigned)q.x & 0xFFu)
         | (((unsigned)q.y & 0xFFu) << 8)
         | (((unsigned)q.z & 0xFFu) << 16)
         | (((unsigned)q.w) << 24);
}

__device__ __forceinline__ void requant_store(const v8i d0, const v8i d1,
                                              float* __restrict__ out,
                                              int obase0, int k0, bool valid) {
    constexpr int RED_M  = (Q_MANTISSA + (1 << 15)) >> 16;
    constexpr int SHIFTS = 15 - EXPONENT;
    if (valid) {
        const int ob0 = obase0 + k0 * DIN;
        const int ob1 = obase0 + (k0 + 2) * DIN;
        #pragma unroll
        for (int v = 0; v < 8; ++v) {
            const long long t0 = (long long)d0[v] * (long long)RED_M
                               + (1LL << (SHIFTS - 1));
            const long long t1 = (long long)d1[v] * (long long)RED_M
                               + (1LL << (SHIFTS - 1));
            int r0 = (int)(t0 >> SHIFTS) + OUT_ZP;
            int r1 = (int)(t1 >> SHIFTS) + OUT_ZP;
            r0 = min(127, max(-128, r0));
            r1 = min(127, max(-128, r1));
            out[ob0 + v * (WOUT * DIN)] = (float)r0;
            out[ob1 + v * (WOUT * DIN)] = (float)r1;
        }
    }
}

__global__ __launch_bounds__(256)
void conv_dw3x3_wmma(const int* __restrict__ x,
                     const int* __restrict__ w,
                     const int* __restrict__ b,
                     float* __restrict__ out)
{
    const int blk  = blockIdx.x;
    const int f    = blk >> 5;
    const int rb   = blk & 31;
    const int j0   = rb << 4;
    const int wv   = threadIdx.x >> 5;
    const int lane = threadIdx.x & 31;
    const int n    = lane & 15;
    const int h    = lane >> 4;

    v8i bm;
    #pragma unroll
    for (int v = 0; v < 8; ++v) {
        unsigned dw = 0;
        #pragma unroll
        for (int byte = 0; byte < 4; ++byte) {
            const int K = ((v >> 1) << 4) + (h << 3) + ((v & 1) << 2) + byte;
            const int m = K >> 4;
            const int c = (K >> 2) & 3;
            const int l = K & 3;
            if (m < 3 && n < 8) {
                const int k  = n >> 2;
                const int lo = n & 3;
                const int t  = c - k;
                if (l == lo && t >= 0 && t <= 2) {
                    dw |= ((unsigned)w[f * 9 + m * 3 + t] & 0xFFu) << (byte * 8);
                }
            }
        }
        bm[v] = (int)dw;
    }

    const int bias = b[f];
    v8i cacc;
    #pragma unroll
    for (int v = 0; v < 8; ++v) cacc[v] = bias;

    const int4* __restrict__ x4 = (const int4*)x;
    int rowb[3];
    #pragma unroll
    for (int m = 0; m < 3; ++m)
        rowb[m] = (f * HIN + j0 + (lane & 15) + m) * WIN + (h << 1);

    const int  obase0 = ((f * HOUT + j0 + (h << 3)) * WOUT) * DIN + n;
    const bool valid  = (n < 8);

    const int ktBeg = wv * 32;

    v8i d0, d1;
    int kprev = ktBeg << 1;
    {
        v8i a0, a1;
        #pragma unroll
        for (int m = 0; m < 3; ++m) {
            #pragma unroll
            for (int ci = 0; ci < 2; ++ci) {
                a0[m * 2 + ci] = (int)pack_i4(x4[rowb[m] + kprev + ci]);
                a1[m * 2 + ci] = (int)pack_i4(x4[rowb[m] + kprev + 2 + ci]);
            }
        }
        a0[6] = 0; a0[7] = 0;
        a1[6] = 0; a1[7] = 0;
        d0 = __builtin_amdgcn_wmma_i32_16x16x64_iu8(true, a0, true, bm, cacc,
                                                    false, false);
        d1 = __builtin_amdgcn_wmma_i32_16x16x64_iu8(true, a1, true, bm, cacc,
                                                    false, false);
    }

    for (int kt = ktBeg + 2; kt < ktBeg + 32; kt += 2) {
        const int k0 = kt << 1;

        v8i a0, a1;
        #pragma unroll
        for (int m = 0; m < 3; ++m) {
            #pragma unroll
            for (int ci = 0; ci < 2; ++ci) {
                a0[m * 2 + ci] = (int)pack_i4(x4[rowb[m] + k0 + ci]);
                a1[m * 2 + ci] = (int)pack_i4(x4[rowb[m] + k0 + 2 + ci]);
            }
        }
        a0[6] = 0; a0[7] = 0;
        a1[6] = 0; a1[7] = 0;

        const v8i e0 = __builtin_amdgcn_wmma_i32_16x16x64_iu8(true, a0, true, bm,
                                                              cacc, false, false);
        const v8i e1 = __builtin_amdgcn_wmma_i32_16x16x64_iu8(true, a1, true, bm,
                                                              cacc, false, false);

        requant_store(d0, d1, out, obase0, kprev, valid);

        d0 = e0;
        d1 = e1;
        kprev = k0;
    }

    requant_store(d0, d1, out, obase0, kprev, valid);
}

extern "C" void kernel_launch(void* const* d_in, const int* in_sizes, int n_in,
                              void* d_out, int out_size, void* d_ws, size_t ws_size,
                              hipStream_t stream) {
    (void)in_sizes; (void)n_in; (void)out_size; (void)d_ws; (void)ws_size;
    const int* x = (const int*)d_in[0];
    const int* w = (const int*)d_in[1];
    const int* b = (const int*)d_in[2];
    float* out   = (float*)d_out;

    dim3 grid(FCH * 32);
    dim3 block(256);
    hipLaunchKernelGGL(conv_dw3x3_wmma, grid, block, 0, stream, x, w, b, out);
}
